// Attention_87806311399904
// MI455X (gfx1250) — matrix operand order repaired and checked by reading, not compiled, not run on hardware
//
#include <hip/hip_runtime.h>
#include <hip/hip_bf16.h>

typedef __attribute__((ext_vector_type(16))) _Float16 v16h;
typedef __attribute__((ext_vector_type(8)))  float    v8f;

union HFrag {
    v16h v;
    unsigned int u[8];
    _Float16 h[16];
};

// CDNA5 async global->LDS DMA (no VGPR bounce, tracked on ASYNCcnt).
// LDS operand = wave-relative LDS address = low 32 bits of generic pointer.
__device__ __forceinline__ void async_copy_b128(const _Float16* gsrc,
                                                _Float16* ldst) {
    unsigned l = (unsigned)(size_t)ldst;
    asm volatile("global_load_async_to_lds_b128 %0, %1, off"
                 :: "v"(l), "v"(gsrc) : "memory");
}
__device__ __forceinline__ void wait_async0() {
    asm volatile("s_wait_asynccnt 0" ::: "memory");
}

// ---------------------------------------------------------------------------
// Conversion / repack kernels (fp32 -> f16)
// ---------------------------------------------------------------------------
__global__ void conv_x_kernel(const float* __restrict__ x,
                              _Float16* __restrict__ xh, int n) {
    int i = blockIdx.x * 256 + threadIdx.x;
    if (i < n) xh[i] = (_Float16)x[i];
}

// Build combined QKV weight as B-matrix [1024 (d_model) x 3072], col c:
//   wt = c/1024 (0=Q,1=K,2=V), h = (c%1024)/64, e = c%64
__global__ void conv_wqkv_kernel(const float* __restrict__ WQ,
                                 const float* __restrict__ WK,
                                 const float* __restrict__ WV,
                                 _Float16* __restrict__ W) {
    int idx = blockIdx.x * 256 + threadIdx.x;   // 1024*3072 exact
    int d = idx / 3072, c = idx % 3072;
    int wt = c >> 10, rem = c & 1023;
    int h = rem >> 6, e = rem & 63;
    const float* src = (wt == 0) ? WQ : ((wt == 1) ? WK : WV);
    W[idx] = (_Float16)src[((size_t)h * 1024 + d) * 64 + e];
}

__global__ void conv_bias_kernel(const float* __restrict__ bQ,
                                 const float* __restrict__ bK,
                                 const float* __restrict__ bV,
                                 float* __restrict__ bias) {
    int c = blockIdx.x * 256 + threadIdx.x;
    if (c >= 3072) return;
    int wt = c >> 10, rem = c & 1023;
    const float* src = (wt == 0) ? bQ : ((wt == 1) ? bK : bV);
    bias[c] = src[rem];
}

__global__ void conv_wo_kernel(const float* __restrict__ WO,
                               _Float16* __restrict__ woh, int n) {
    int i = blockIdx.x * 256 + threadIdx.x;
    if (i < n) woh[i] = (_Float16)WO[i];   // already [(h*64+e) x 1024] row-major
}

// ---------------------------------------------------------------------------
// Generic f16 GEMM:  C[MxN] = A[MxK] * B[KxN] + bias
//   mode 0 -> store f16 to Ch,  mode 1 -> store f32 to Cf
// Block: 128 threads (4 waves), 128x64 output tile, K-step 32.
// Wave computes 32x64 (2 A-frags x 4 B-frags = 8 WMMAs / K-step).
// A tile copied via async DMA (global_load_async_to_lds_b128);
// B tile loaded to regs then stored transposed so frag reads merge to b128.
// ---------------------------------------------------------------------------
__global__ __launch_bounds__(128) void gemm_f16_kernel(
    const _Float16* __restrict__ A, const _Float16* __restrict__ B,
    const float* __restrict__ bias, _Float16* __restrict__ Ch,
    float* __restrict__ Cf, int M, int N, int K, int mode) {
    __shared__ _Float16 As[128 * 32];   // [row][k]
    __shared__ _Float16 BsT[64 * 32];   // [col][k]

    const int tid  = threadIdx.x;
    const int wave = tid >> 5;
    const int lane = tid & 31;
    const int half = lane >> 4;
    const int ln16 = lane & 15;
    const int m0 = blockIdx.x * 128;
    const int n0 = blockIdx.y * 64;

    v8f acc[2][4] = {};

    for (int k0 = 0; k0 < K; k0 += 32) {
        // B tile (32x64): issue both global loads first (overlapped)
        float4 vB[2];
#pragma unroll
        for (int it = 0; it < 2; ++it) {
            int idx = tid + it * 128;
            int rb = idx >> 3, cbb = (idx & 7) << 3;
            vB[it] = *(const float4*)(B + (size_t)(k0 + rb) * N + n0 + cbb);
        }
        // A tile (128x32): async DMA straight into LDS, 4 x b128 per lane
#pragma unroll
        for (int it = 0; it < 4; ++it) {
            int idx = tid + it * 128;
            int r = idx >> 2, cb = (idx & 3) << 3;
            async_copy_b128(A + (size_t)(m0 + r) * K + k0 + cb,
                            As + r * 32 + cb);
        }
        // scatter-store B transposed
#pragma unroll
        for (int it = 0; it < 2; ++it) {
            int idx = tid + it * 128;
            int rb = idx >> 3, cbb = (idx & 7) << 3;
            const _Float16* pb = (const _Float16*)&vB[it];
#pragma unroll
            for (int j = 0; j < 8; ++j) BsT[(cbb + j) * 32 + rb] = pb[j];
        }
        // prefetch next K-tile lines while this one computes
        int kn = k0 + 32;
        if (kn < K) {
            __builtin_prefetch(A + (size_t)(m0 + tid) * K + kn, 0, 1);
            if (tid < 32)
                __builtin_prefetch(B + (size_t)(kn + tid) * N + n0, 0, 1);
        }
        wait_async0();
        __syncthreads();

        // A-frags (16x32 f16): lane m=lane%16; i<4: k=half*8+2i, i>=4: 16+half*8+2(i-4)
        HFrag a[2];
#pragma unroll
        for (int af = 0; af < 2; ++af) {
            int m = 32 * wave + af * 16 + ln16;
#pragma unroll
            for (int i = 0; i < 8; ++i) {
                int k = (i < 4) ? (half * 8 + i * 2) : (16 + half * 8 + (i - 4) * 2);
                a[af].u[i] = *(const unsigned int*)(As + m * 32 + k);
            }
        }
#pragma unroll
        for (int nb = 0; nb < 4; ++nb) {
            // B-frag (32x16): lane col n=lane%16; i<4: krow=half*8+2i, i>=4: 16+half*8+2(i-4)
            HFrag bf;
#pragma unroll
            for (int i = 0; i < 8; ++i) {
                int kr = (i < 4) ? (half * 8 + i * 2) : (16 + half * 8 + (i - 4) * 2);
                bf.u[i] = *(const unsigned int*)(BsT + (nb * 16 + ln16) * 32 + kr);
            }
#pragma unroll
            for (int af = 0; af < 2; ++af)
                acc[af][nb] = __builtin_amdgcn_wmma_f32_16x16x32_f16(
                    false, a[af].v, false, bf.v, (short)0, acc[af][nb], false, false);
        }
        __syncthreads();
    }

    // epilogue: C-frag row m = i + half*8, col n = lane%16
#pragma unroll
    for (int af = 0; af < 2; ++af) {
#pragma unroll
        for (int nb = 0; nb < 4; ++nb) {
            int gn = n0 + nb * 16 + ln16;
            float bv = bias ? bias[gn] : 0.0f;
#pragma unroll
            for (int i = 0; i < 8; ++i) {
                int gm = m0 + 32 * wave + af * 16 + half * 8 + i;
                float v = acc[af][nb][i] + bv;
                if (mode == 0) Ch[(size_t)gm * N + gn] = (_Float16)v;
                else           Cf[(size_t)gm * N + gn] = v;
            }
        }
    }
}

// ---------------------------------------------------------------------------
// Flash attention. Grid: (S/64, H, B). Block: 128 threads = 4 waves.
// Wave w owns query rows [qt*64 + 16w, +16). Key tiles of 32, staged in LDS.
// QKV layout: [b*2048+s][3072] with Q at col h*64, K at 1024+h*64, V at 2048+h*64.
// Output Zh: [b*2048+q][1024] f16 (col h*64+e).
// ---------------------------------------------------------------------------
__global__ __launch_bounds__(128) void attn_kernel(
    const _Float16* __restrict__ QKV, _Float16* __restrict__ Zh) {
    __shared__ _Float16 lk[32 * 64];     // K tile   [key][e]
    __shared__ _Float16 lvT[64 * 32];    // V tile T [e][key]
    __shared__ _Float16 sp[4][16 * 32];  // per-wave P scratch [m][k]

    const int tid  = threadIdx.x;
    const int wave = tid >> 5;
    const int lane = tid & 31;
    const int half = lane >> 4;
    const int ln16 = lane & 15;
    const int qt = blockIdx.x, h = blockIdx.y, b = blockIdx.z;
    const int qbase = qt * 64;
    const int qw = qbase + wave * 16;

    const _Float16* Qp = QKV + (size_t)b * 2048 * 3072 + h * 64;
    const _Float16* Kp = Qp + 1024;
    const _Float16* Vp = Qp + 2048;

    // Q A-frags for e=[0,32) and e=[32,64)
    HFrag qa[2];
    {
        const _Float16* qrow = Qp + (size_t)(qw + ln16) * 3072;
#pragma unroll
        for (int eb = 0; eb < 2; ++eb)
#pragma unroll
            for (int i = 0; i < 8; ++i) {
                int k = (i < 4) ? (half * 8 + i * 2) : (16 + half * 8 + (i - 4) * 2);
                qa[eb].u[i] = *(const unsigned int*)(qrow + eb * 32 + k);
            }
    }

    v8f zacc[4] = {};
    float mrun[8], lrun[8];
#pragma unroll
    for (int i = 0; i < 8; ++i) { mrun[i] = -1.0e30f; lrun[i] = 0.0f; }

    const float scale = 0.125f;              // 1/sqrt(64)
    const int ntiles = (qbase + 64) / 32;    // keys [0, qbase+64)

    for (int jt = 0; jt < ntiles; ++jt) {
        const int jb = jt * 32;
        // V tile (32x64): issue both global loads first (overlapped)
        float4 vV[2];
#pragma unroll
        for (int it = 0; it < 2; ++it) {
            int idx = tid + it * 128;
            int key = idx >> 3, cb = (idx & 7) << 3;
            vV[it] = *(const float4*)(Vp + (size_t)(jb + key) * 3072 + cb);
        }
        // K tile (32x64) row-major: async DMA straight into LDS
#pragma unroll
        for (int it = 0; it < 2; ++it) {
            int idx = tid + it * 128;
            int key = idx >> 3, cb = (idx & 7) << 3;
            async_copy_b128(Kp + (size_t)(jb + key) * 3072 + cb,
                            lk + key * 64 + cb);
        }
        // scatter-store V transposed [e][key]
#pragma unroll
        for (int it = 0; it < 2; ++it) {
            int idx = tid + it * 128;
            int key = idx >> 3, cb = (idx & 7) << 3;
            const _Float16* pv = (const _Float16*)&vV[it];
#pragma unroll
            for (int j = 0; j < 8; ++j) lvT[(cb + j) * 32 + key] = pv[j];
        }
        // prefetch next key tile (K and V rows; 32 lines each)
        if (jt + 1 < ntiles && tid < 32) {
            __builtin_prefetch(Kp + (size_t)(jb + 32 + tid) * 3072, 0, 1);
            __builtin_prefetch(Vp + (size_t)(jb + 32 + tid) * 3072, 0, 1);
        }
        wait_async0();
        __syncthreads();

        if (jb <= qw + 15) {   // wave has unmasked work in this key tile
            // scores: 16x32 via two 16x16 C-frags, K-loop over e (2x32)
            v8f s[2] = {};
#pragma unroll
            for (int eb = 0; eb < 2; ++eb) {
#pragma unroll
                for (int nb = 0; nb < 2; ++nb) {
                    HFrag bk;   // B = K^T: rows=e, cols=key  -> lk[key][e]
#pragma unroll
                    for (int i = 0; i < 8; ++i) {
                        int er = eb * 32 + ((i < 4) ? (half * 8 + i * 2) : (16 + half * 8 + (i - 4) * 2));
                        bk.u[i] = *(const unsigned int*)(lk + (nb * 16 + ln16) * 64 + er);
                    }
                    s[nb] = __builtin_amdgcn_wmma_f32_16x16x32_f16(
                        false, qa[eb].v, false, bk.v, (short)0, s[nb], false, false);
                }
            }

            // mask + online softmax (row = i + half*8, spread over 16 lanes)
#pragma unroll
            for (int i = 0; i < 8; ++i) {
                int q = qw + half * 8 + i;
                float s0 = s[0][i] * scale;
                float s1 = s[1][i] * scale;
                if (jb + ln16 > q)      s0 = -100000.0f;   // reference MASK_FILL
                if (jb + 16 + ln16 > q) s1 = -100000.0f;
                float rmax = fmaxf(s0, s1);
#pragma unroll
                for (int off = 1; off < 16; off <<= 1)
                    rmax = fmaxf(rmax, __shfl_xor(rmax, off, 32));
                float mnew  = fmaxf(mrun[i], rmax);
                float p0    = __expf(s0 - mnew);
                float p1    = __expf(s1 - mnew);
                float alpha = __expf(mrun[i] - mnew);
                float rsum  = p0 + p1;
#pragma unroll
                for (int off = 1; off < 16; off <<= 1)
                    rsum += __shfl_xor(rsum, off, 32);
                lrun[i] = lrun[i] * alpha + rsum;
                mrun[i] = mnew;
#pragma unroll
                for (int nb = 0; nb < 4; ++nb) zacc[nb][i] *= alpha;
                int m = half * 8 + i;
                sp[wave][m * 32 + ln16]      = (_Float16)p0;
                sp[wave][m * 32 + 16 + ln16] = (_Float16)p1;
            }
            // wave-private LDS round trip: C-frag layout -> A-frag layout
            asm volatile("s_wait_dscnt 0" ::: "memory");
            HFrag pa;
#pragma unroll
            for (int i = 0; i < 8; ++i) {
                int k = (i < 4) ? (half * 8 + i * 2) : (16 + half * 8 + (i - 4) * 2);
                pa.u[i] = *(const unsigned int*)(&sp[wave][ln16 * 32 + k]);
            }
            // Z += P(16x32) * V(32x64): 4 WMMAs over e-blocks of 16
#pragma unroll
            for (int nb = 0; nb < 4; ++nb) {
                HFrag bv;   // B = V: rows=key, cols=e -> lvT[e][key]
#pragma unroll
                for (int i = 0; i < 8; ++i) {
                    int kr = (i < 4) ? (half * 8 + i * 2) : (16 + half * 8 + (i - 4) * 2);
                    bv.u[i] = *(const unsigned int*)(lvT + (nb * 16 + ln16) * 32 + kr);
                }
                zacc[nb] = __builtin_amdgcn_wmma_f32_16x16x32_f16(
                    false, pa.v, false, bv.v, (short)0, zacc[nb], false, false);
            }
        }
        __syncthreads();
    }

    // finalize: Z = acc / l, store f16 into Zh[b*2048+q][h*64+e]
    _Float16* Zbase = Zh + (size_t)b * 2048 * 1024 + h * 64;
#pragma unroll
    for (int i = 0; i < 8; ++i) {
        float inv = 1.0f / lrun[i];
        int q = qw + half * 8 + i;
        _Float16* zr = Zbase + (size_t)q * 1024;
#pragma unroll
        for (int nb = 0; nb < 4; ++nb)
            zr[nb * 16 + ln16] = (_Float16)(zacc[nb][i] * inv);
    }
}

// ---------------------------------------------------------------------------
// Host launcher
// ---------------------------------------------------------------------------
extern "C" void kernel_launch(void* const* d_in, const int* in_sizes, int n_in,
                              void* d_out, int out_size, void* d_ws, size_t ws_size,
                              hipStream_t stream) {
    (void)in_sizes; (void)n_in; (void)out_size; (void)ws_size;
    const float* x  = (const float*)d_in[0];
    const float* WQ = (const float*)d_in[1];
    const float* WK = (const float*)d_in[2];
    const float* WV = (const float*)d_in[3];
    const float* WO = (const float*)d_in[4];
    const float* bQ = (const float*)d_in[5];
    const float* bK = (const float*)d_in[6];
    const float* bV = (const float*)d_in[7];
    const float* bO = (const float*)d_in[8];

    char* ws = (char*)d_ws;
    _Float16* xh    = (_Float16*)(ws);                       //  8 MB: x f16 [4096][1024]
    _Float16* wqkv  = (_Float16*)(ws + (8u  << 20));         //  6 MB: [1024][3072]
    _Float16* woh   = (_Float16*)(ws + (14u << 20));         //  2 MB: [1024][1024]
    _Float16* qkv   = (_Float16*)(ws + (16u << 20));         // 24 MB: [4096][3072]
    _Float16* zh    = (_Float16*)(ws + (40u << 20));         //  8 MB: [4096][1024]
    float*    biasq = (float*)   (ws + (48u << 20));         // 12 KB: [3072]

    conv_x_kernel   <<<(4096 * 1024) / 256, 256, 0, stream>>>(x, xh, 4096 * 1024);
    conv_wqkv_kernel<<<(1024 * 3072) / 256, 256, 0, stream>>>(WQ, WK, WV, wqkv);
    conv_bias_kernel<<<12, 256, 0, stream>>>(bQ, bK, bV, biasq);
    conv_wo_kernel  <<<(1024 * 1024) / 256, 256, 0, stream>>>(WO, woh, 1024 * 1024);

    // QKV projection: [4096x1024] * [1024x3072] -> f16
    gemm_f16_kernel<<<dim3(32, 48), 128, 0, stream>>>(
        xh, wqkv, biasq, qkv, nullptr, 4096, 3072, 1024, 0);

    // Flash attention per (qtile, head, batch)
    attn_kernel<<<dim3(32, 16, 2), 128, 0, stream>>>(qkv, zh);

    // Output projection: [4096x1024] * [1024x1024] + b_O -> f32 d_out
    gemm_f16_kernel<<<dim3(32, 16), 128, 0, stream>>>(
        zh, woh, bO, nullptr, (float*)d_out, 4096, 1024, 1024, 1);
}
